// LagrangianNeuralNetworkLayer_46179488366778
// MI455X (gfx1250) — matrix operand order repaired and checked by reading, not compiled, not run on hardware
//
#include <hip/hip_runtime.h>

typedef __attribute__((ext_vector_type(16))) _Float16 v16h;
typedef __attribute__((ext_vector_type(8)))  _Float16 v8h;
typedef __attribute__((ext_vector_type(8)))  float    v8f;

#define DD   16
#define XX   32
#define N1   256
#define N2   256
#define N3   128
#define STR0 264   // padded row stride (halfs) for 32x256 tangent buffer
#define STR1 136   // padded row stride (halfs) for 32x128 tangent buffer
#define SWL  264   // padded row stride (halfs) for LDS weight panels

// ---- per-sample LDS slice layout (bytes) ----
#define OFF_TB0 0                       // 32*264*2 = 16896  raw f16 A2_dot
#define OFF_TB1 16896                   // 32*136*2 = 8704   raw f16 A3_dot
#define OFF_SH1 25600                   // f16 sigma'(a1) [256]
#define OFF_SH2 (OFF_SH1 + 512)
#define OFF_SH3 (OFF_SH2 + 512)        // [128]
#define OFF_DH1 (OFF_SH3 + 256)        // f16 sigma''(a)*u [256]
#define OFF_DH2 (OFF_DH1 + 512)
#define OFF_DH3 (OFF_DH2 + 512)        // [128]
#define OFF_H1F (OFF_DH3 + 256)        // f32 h1 (then g1) [256]
#define OFF_H2F (OFF_H1F + 1024)       // f32 h2 (then g2) [256]
#define OFF_XS  (OFF_H2F + 1024)       // f32 x [32]
#define OFF_GX  (OFF_XS  + 128)        // f32 grad_x [32] (reused as solution y)
#define PER_SAMPLE (OFF_GX + 128)      // 30464 bytes
// time-multiplexed aliases (regions dead when these go live):
#define OFF_HS  OFF_TB0                // f32 H[16][34]  (tb0 dead after chain3)
#define OFF_AUG (OFF_TB0 + 2176)       // f32 aug[16][18]
#define OFF_GG  OFF_TB1                // f32 g3 [128]   (tb1 written only in chain3)

#define NWAVE 4
#define OFF_W2L ((size_t)NWAVE * PER_SAMPLE)          // 121856, 256*264*2 = 135168
#define OFF_W3L (OFF_W2L + (size_t)N1 * SWL * 2)      // 257024, 128*264*2 = 67584
#define SMEM_TOTAL (OFF_W3L + (size_t)N3 * SWL * 2)   // 324608 <= 327680 (320KB)

__device__ __forceinline__ float sigmoidf_(float a) { return 1.0f / (1.0f + __expf(-a)); }
__device__ __forceinline__ float softplusf_(float a) { return (a > 20.0f) ? a : log1pf(__expf(a)); }

// A operand (16x32 f16): lane L<16 holds row m0+L, K chunks [k0..k0+7],[k0+16..k0+23];
// lanes 16..31 same rows, chunks +8. Matches documented 16-bit A layout.
template <bool SCALE>
__device__ __forceinline__ v16h load_a16(const _Float16* buf, int rs, int m0, int k0,
                                         const _Float16* scale) {
  const int lane = threadIdx.x & 31;
  const int m  = m0 + (lane & 15);
  const int kb = k0 + ((lane >> 4) << 3);
  union { v16h v; v8h h[2]; } r;
  r.h[0] = *(const v8h*)(buf + (size_t)m * rs + kb);
  r.h[1] = *(const v8h*)(buf + (size_t)m * rs + kb + 16);
  if constexpr (SCALE) {
    r.h[0] *= *(const v8h*)(scale + kb);
    r.h[1] *= *(const v8h*)(scale + kb + 16);
  }
  return r.v;
}

// B operand (32x16 f16): lane L holds column n0+(L&15), K chunks [k0..k0+7],[k0+16..k0+23]
// (+8 for lanes 16..31). Source buffer layout is [n][k] row-major.
__device__ __forceinline__ v16h load_b16(const _Float16* buf, int rs, int n0, int k0) {
  const int lane = threadIdx.x & 31;
  const int n  = n0 + (lane & 15);
  const int kb = k0 + ((lane >> 4) << 3);
  union { v16h v; v8h h[2]; } r;
  r.h[0] = *(const v8h*)(buf + (size_t)n * rs + kb);
  r.h[1] = *(const v8h*)(buf + (size_t)n * rs + kb + 16);
  return r.v;
}

__device__ __forceinline__ v8f wmma_f16(v16h a, v16h b, v8f c) {
  return __builtin_amdgcn_wmma_f32_16x16x32_f16(false, a, false, b, (short)0, c, false, false);
}

// D tile (f32 16x16): lane L, VGPR v -> (m = v + 8*(L>=16), n = L&15). Store as f16 [m][n].
__device__ __forceinline__ void store_d_f16(v8f acc, _Float16* buf, int rs, int m0, int n0) {
  const int lane = threadIdx.x & 31;
  const int n  = n0 + (lane & 15);
  const int mh = m0 + ((lane >> 4) << 3);
#pragma unroll
  for (int v = 0; v < 8; ++v) buf[(size_t)(mh + v) * rs + n] = (_Float16)acc[v];
}

// one-time weight prep: W1 f32 -> f16 (layout preserved [in][out])
__global__ void lnn_prep_kernel(const float* __restrict__ W1, _Float16* __restrict__ W1h) {
  int id = blockIdx.x * blockDim.x + threadIdx.x;
  int st = gridDim.x * blockDim.x;
  for (int e = id; e < XX * N1; e += st) W1h[e] = (_Float16)W1[e];
}

__global__ void __launch_bounds__(128, 1)
lnn_main_kernel(const float* __restrict__ q, const float* __restrict__ qd,
                const float* __restrict__ b1, const float* __restrict__ b2,
                const float* __restrict__ b3, const float* __restrict__ W4,
                const float* __restrict__ W2, const float* __restrict__ W3,
                const _Float16* __restrict__ W1h, float* __restrict__ out, int Bn) {
  extern __shared__ __attribute__((aligned(16))) char smem[];
  const int lane = threadIdx.x & 31;
  const int wv   = threadIdx.x >> 5;
  char* sb = smem + (size_t)wv * PER_SAMPLE;
  _Float16* tb0 = (_Float16*)(sb + OFF_TB0);
  _Float16* tb1 = (_Float16*)(sb + OFF_TB1);
  _Float16* sh1 = (_Float16*)(sb + OFF_SH1);
  _Float16* sh2 = (_Float16*)(sb + OFF_SH2);
  _Float16* sh3 = (_Float16*)(sb + OFF_SH3);
  _Float16* dh1 = (_Float16*)(sb + OFF_DH1);
  _Float16* dh2 = (_Float16*)(sb + OFF_DH2);
  _Float16* dh3 = (_Float16*)(sb + OFF_DH3);
  float* h1f = (float*)(sb + OFF_H1F);
  float* h2f = (float*)(sb + OFF_H2F);
  float* xs  = (float*)(sb + OFF_XS);
  float* gx  = (float*)(sb + OFF_GX);
  float* gg  = (float*)(sb + OFF_GG);   // aliases tb1 (dead until chain3)
  float* Hs  = (float*)(sb + OFF_HS);   // aliases tb0 (dead after chain3)
  float* aug = (float*)(sb + OFF_AUG);  // aliases tb0
  _Float16* w2l = (_Float16*)(smem + OFF_W2L);  // [out 256][in 256] stride SWL
  _Float16* w3l = (_Float16*)(smem + OFF_W3L);  // [out 128][in 256] stride SWL

  // ---- stage shared weight panels into LDS (once per block) ----
  __builtin_prefetch(W1h, 0, 3);
  for (int e = threadIdx.x; e < N1 * N2; e += blockDim.x) {
    int i = e >> 8, j = e & 255;                 // W2[i][j], i=in, j=out
    w2l[(size_t)j * SWL + i] = (_Float16)W2[e];
  }
  for (int e = threadIdx.x; e < N2 * N3; e += blockDim.x) {
    int i = e >> 7, j = e & 127;                 // W3[i][j]
    w3l[(size_t)j * SWL + i] = (_Float16)W3[e];
  }
  __syncthreads();

  const int nwv = gridDim.x * (blockDim.x >> 5);
  const int wid = blockIdx.x * (blockDim.x >> 5) + wv;
  const int iters = (Bn + nwv - 1) / nwv;

  for (int it = 0; it < iters; ++it) {
    int s = wid + it * nwv;
    if (s >= Bn) s = Bn - 1;  // duplicate last sample: deterministic, same output

    // ---- x = concat(q, q_dot) ----
    xs[lane] = (lane < DD) ? q[(size_t)s * DD + lane] : qd[(size_t)s * DD + lane - DD];
    __syncthreads();

    // ---- forward L1 (W1 from global f16, coalesced across lanes) ----
    for (int r = 0; r < N1 / 32; ++r) {
      int u = lane + 32 * r;
      float a = b1[u];
#pragma unroll 8
      for (int i = 0; i < XX; ++i) a = fmaf(xs[i], (float)W1h[i * N1 + u], a);
      h1f[u] = softplusf_(a);
      sh1[u] = (_Float16)sigmoidf_(a);
    }
    __syncthreads();

    // ---- forward L2 (row-contiguous v8h loads from LDS) ----
    for (int r = 0; r < N2 / 32; ++r) {
      int u = lane + 32 * r;
      float a = b2[u];
      const _Float16* wr = w2l + (size_t)u * SWL;
#pragma unroll 4
      for (int c = 0; c < N1 / 8; ++c) {
        v8h w = *(const v8h*)(wr + c * 8);
#pragma unroll
        for (int e = 0; e < 8; ++e) a = fmaf(h1f[c * 8 + e], (float)w[e], a);
      }
      h2f[u] = softplusf_(a);
      sh2[u] = (_Float16)sigmoidf_(a);
    }
    __syncthreads();

    // ---- forward L3 + u3 = W4 ----
    for (int r = 0; r < N3 / 32; ++r) {
      int u = lane + 32 * r;
      float a = b3[u];
      const _Float16* wr = w3l + (size_t)u * SWL;
#pragma unroll 4
      for (int c = 0; c < N2 / 8; ++c) {
        v8h w = *(const v8h*)(wr + c * 8);
#pragma unroll
        for (int e = 0; e < 8; ++e) a = fmaf(h2f[c * 8 + e], (float)w[e], a);
      }
      float sg = sigmoidf_(a);
      float w4 = W4[u];
      sh3[u] = (_Float16)sg;
      gg[u]  = sg * w4;                            // g3
      dh3[u] = (_Float16)(sg * (1.0f - sg) * w4);  // d3
    }
    __syncthreads();

    // ---- reverse: u2 = W3 g3 (lane owns contiguous 8-wide input chunk) ----
    {
      const int i0 = lane * 8;
      float acc[8] = {0, 0, 0, 0, 0, 0, 0, 0};
      for (int j = 0; j < N3; ++j) {
        float g = gg[j];  // broadcast
        v8h w = *(const v8h*)(w3l + (size_t)j * SWL + i0);
#pragma unroll
        for (int e = 0; e < 8; ++e) acc[e] = fmaf((float)w[e], g, acc[e]);
      }
      __syncthreads();
#pragma unroll
      for (int e = 0; e < 8; ++e) {
        float sg = (float)sh2[i0 + e];
        h2f[i0 + e] = sg * acc[e];                              // g2
        dh2[i0 + e] = (_Float16)(sg * (1.0f - sg) * acc[e]);
      }
    }
    __syncthreads();

    // ---- reverse: u1 = W2 g2 ----
    {
      const int i0 = lane * 8;
      float acc[8] = {0, 0, 0, 0, 0, 0, 0, 0};
      for (int j = 0; j < N2; ++j) {
        float g = h2f[j];  // broadcast
        v8h w = *(const v8h*)(w2l + (size_t)j * SWL + i0);
#pragma unroll
        for (int e = 0; e < 8; ++e) acc[e] = fmaf((float)w[e], g, acc[e]);
      }
#pragma unroll
      for (int e = 0; e < 8; ++e) {
        float sg = (float)sh1[i0 + e];
        h1f[i0 + e] = sg * acc[e];                              // g1
        dh1[i0 + e] = (_Float16)(sg * (1.0f - sg) * acc[e]);
      }
    }
    __syncthreads();

    // ---- grad_x = W1 g1 ----
    {
      float a = 0.0f;
      const _Float16* wr = W1h + (size_t)lane * N1;
#pragma unroll 4
      for (int c = 0; c < N1 / 8; ++c) {
        v8h w = *(const v8h*)(wr + c * 8);
#pragma unroll
        for (int e = 0; e < 8; ++e) a = fmaf((float)w[e], h1f[c * 8 + e], a);
      }
      gx[lane] = a;
    }
    __syncthreads();

    // ================= tangent chain + Hessian via WMMA =================
    v8f Hc0 = {};  // H(qd rows, cols 0..15)  = mixed block
    v8f Hc1 = {};  // H(qd rows, cols 16..31) = hess_qd

    // layer 1: Adot1 = W1 rows; depth-1 software pipeline over kt (global W1h)
    {
      v16h a  = load_a16<true>(W1h, N1, 16, 0, dh1);
      v16h bA = load_b16(W1h, N1, 0, 0);
      v16h bB = load_b16(W1h, N1, 16, 0);
#pragma unroll
      for (int kt = 0; kt < 8; ++kt) {
        v16h an = a, bAn = bA, bBn = bB;
        if (kt < 7) {
          an  = load_a16<true>(W1h, N1, 16, (kt + 1) * 32, dh1);
          bAn = load_b16(W1h, N1, 0,  (kt + 1) * 32);
          bBn = load_b16(W1h, N1, 16, (kt + 1) * 32);
        }
        Hc0 = wmma_f16(a, bA, Hc0);
        Hc1 = wmma_f16(a, bB, Hc1);
        a = an; bA = bAn; bB = bBn;
      }
    }

    // chain: tb0 = (sigma'1 .* W1) @ W2   [32 x 256], B panel from LDS
    for (int mt = 0; mt < 2; ++mt) {
      v16h Ar[8];
#pragma unroll
      for (int kt = 0; kt < 8; ++kt) Ar[kt] = load_a16<true>(W1h, N1, mt * 16, kt * 32, sh1);
      for (int nt = 0; nt < N2 / 16; ++nt) {
        v16h Br[8];
#pragma unroll
        for (int kt = 0; kt < 8; ++kt) Br[kt] = load_b16(w2l, SWL, nt * 16, kt * 32);
        v8f acc = {};
#pragma unroll
        for (int kt = 0; kt < 8; ++kt) acc = wmma_f16(Ar[kt], Br[kt], acc);
        store_d_f16(acc, tb0, STR0, mt * 16, nt * 16);
      }
    }
    __syncthreads();

    // layer 2 Hessian contraction (operands from LDS tb0)
#pragma unroll
    for (int kt = 0; kt < 8; ++kt) {
      v16h a  = load_a16<true>(tb0, STR0, 16, kt * 32, dh2);
      v16h bA = load_b16(tb0, STR0, 0,  kt * 32);
      v16h bB = load_b16(tb0, STR0, 16, kt * 32);
      Hc0 = wmma_f16(a, bA, Hc0);
      Hc1 = wmma_f16(a, bB, Hc1);
    }

    // chain: tb1 = (sigma'2 .* Adot2) @ W3   [32 x 128]
    for (int mt = 0; mt < 2; ++mt) {
      v16h Ar[8];
#pragma unroll
      for (int kt = 0; kt < 8; ++kt) Ar[kt] = load_a16<true>(tb0, STR0, mt * 16, kt * 32, sh2);
      for (int nt = 0; nt < N3 / 16; ++nt) {
        v16h Br[8];
#pragma unroll
        for (int kt = 0; kt < 8; ++kt) Br[kt] = load_b16(w3l, SWL, nt * 16, kt * 32);
        v8f acc = {};
#pragma unroll
        for (int kt = 0; kt < 8; ++kt) acc = wmma_f16(Ar[kt], Br[kt], acc);
        store_d_f16(acc, tb1, STR1, mt * 16, nt * 16);
      }
    }
    __syncthreads();

    // layer 3 Hessian contraction (K = 128)
#pragma unroll
    for (int kt = 0; kt < 4; ++kt) {
      v16h a  = load_a16<true>(tb1, STR1, 16, kt * 32, dh3);
      v16h bA = load_b16(tb1, STR1, 0,  kt * 32);
      v16h bB = load_b16(tb1, STR1, 16, kt * 32);
      Hc0 = wmma_f16(a, bA, Hc0);
      Hc1 = wmma_f16(a, bB, Hc1);
    }

    // spill H (qd rows, 32 cols) to f32 LDS (aliases dead tb0)
    {
      int n  = lane & 15;
      int mh = (lane >> 4) << 3;
#pragma unroll
      for (int v = 0; v < 8; ++v) {
        Hs[(mh + v) * 34 + n]      = Hc0[v];
        Hs[(mh + v) * 34 + 16 + n] = Hc1[v];
      }
    }
    __syncthreads();

    // ================= per-sample 16x16 solve =================
    if (lane < DD) {
      float f = gx[lane];
#pragma unroll
      for (int j = 0; j < DD; ++j) f -= Hs[lane * 34 + j] * xs[DD + j];
      aug[lane * 18 + 16] = f;
    }
    for (int idx = lane; idx < DD * DD; idx += 32) {
      int r = idx >> 4, c = idx & 15;
      aug[r * 18 + c] = Hs[r * 34 + 16 + c] + ((r == c) ? 1e-6f : 0.0f);
    }
    __syncthreads();

    for (int p = 0; p < DD; ++p) {
      float inv = 1.0f / aug[p * 18 + p];
      for (int r = p + 1; r < DD; ++r) {
        float f = aug[r * 18 + p] * inv;
        int c = p + lane;
        if (c < 17) aug[r * 18 + c] -= f * aug[p * 18 + c];
      }
      __syncthreads();
    }
    for (int p = DD - 1; p >= 0; --p) {
      float yp = aug[p * 18 + 16] / aug[p * 18 + p];
      if (lane < p) aug[lane * 18 + 16] -= aug[lane * 18 + p] * yp;
      if (lane == 0) gx[p] = yp;
      __syncthreads();
    }

    if (lane < DD) out[(size_t)s * DD + lane] = gx[lane];
    __syncthreads();
  }
}

extern "C" void kernel_launch(void* const* d_in, const int* in_sizes, int n_in,
                              void* d_out, int out_size, void* d_ws, size_t ws_size,
                              hipStream_t stream) {
  (void)n_in; (void)out_size; (void)ws_size;
  const float* q  = (const float*)d_in[0];
  const float* qd = (const float*)d_in[1];
  const float* W1 = (const float*)d_in[2];
  const float* b1 = (const float*)d_in[3];
  const float* W2 = (const float*)d_in[4];
  const float* b2 = (const float*)d_in[5];
  const float* W3 = (const float*)d_in[6];
  const float* b3 = (const float*)d_in[7];
  const float* W4 = (const float*)d_in[8];
  // d_in[9] = b4: additive constant, no effect on any derivative

  const int Bn = in_sizes[0] / DD;

  _Float16* W1h = (_Float16*)d_ws;  // 32*256 f16 = 16KB

  lnn_prep_kernel<<<32, 256, 0, stream>>>(W1, W1h);

  const size_t smem = SMEM_TOTAL;  // 324608 B (<= 320KB WGP limit)
  (void)hipFuncSetAttribute(reinterpret_cast<const void*>(lnn_main_kernel),
                            hipFuncAttributeMaxDynamicSharedMemorySize, (int)smem);
  lnn_main_kernel<<<512, 128, smem, stream>>>(q, qd, b1, b2, b3, W4, W2, W3, W1h,
                                              (float*)d_out, Bn);
}
